// Encoder_54090818126872
// MI455X (gfx1250) — matrix operand order repaired and checked by reading, not compiled, not run on hardware
//
#include <hip/hip_runtime.h>
#include <hip/hip_bf16.h>
#include <math.h>

#define DEV __device__ __forceinline__

typedef __attribute__((ext_vector_type(16))) _Float16 v16h;
typedef __attribute__((ext_vector_type(8)))  _Float16 h8;
typedef __attribute__((ext_vector_type(8)))  float    v8f;
typedef __attribute__((ext_vector_type(4)))  int      v4i;

// ---- model dims ----
#define BATCH 4
#define SEQ   1024
#define DM    512
#define NH    8
#define DK    64
#define FF    2048
#define NLAY  6
#define ROWS  (BATCH*SEQ)   // 4096

// ---------------- CDNA5 async-to-LDS support (guarded) ----------------
#if defined(__has_builtin)
#  if __has_builtin(__builtin_amdgcn_global_load_async_to_lds_b128) && \
      __has_builtin(__builtin_amdgcn_s_wait_asynccnt)
#    define USE_ASYNC 1
#  endif
#endif

typedef __attribute__((address_space(1))) v4i gv4i;   // global v4i
typedef __attribute__((address_space(3))) v4i lv4i;   // LDS v4i

DEV void async_cp16(const _Float16* g, _Float16* l){
#if defined(USE_ASYNC)
  __builtin_amdgcn_global_load_async_to_lds_b128((gv4i*)g, (lv4i*)l, 0, 0);
#else
  *(h8*)l = *(const h8*)g;
#endif
}

DEV void async_wait(){
#if defined(USE_ASYNC)
  __builtin_amdgcn_s_wait_asynccnt(0);
#endif
}

// ---------------- WMMA fragment helpers (16x16x32 f16, wave32) ----------------
// A: 16x32 (MxK) row-major; per ISA layout lo-lanes hold K{0..7,16..23},
// hi-lanes K{8..15,24..31} -> two contiguous 8-half (16B) loads per lane.
DEV v16h load_a_h16(const _Float16* A, int ld, int row, int k0, int hi){
  const _Float16* p = A + (size_t)row*ld + k0 + (hi ? 8 : 0);
  h8 lo = *(const h8*)(p);
  h8 hh = *(const h8*)(p + 16);
  v16h a;
#pragma unroll
  for(int i=0;i<8;i++){ a[i]=lo[i]; a[i+8]=hh[i]; }
  return a;
}

DEV v16h load_a_f32(const float* A, int ld, int row, int k0, int hi){
  const float* p = A + (size_t)row*ld + k0 + (hi ? 8 : 0);
  v16h a;
#pragma unroll
  for(int i=0;i<8;i++){ a[i]=(_Float16)p[i]; a[i+8]=(_Float16)p[i+16]; }
  return a;
}

// B[k][n] = Bt[n][k], Bt row-major [N x K]: lo-lanes K{0..7,16..23}, hi-lanes K{8..15,24..31}
DEV v16h load_b_tr(const _Float16* Bt, int ld, int k0, int col, int hi){
  const _Float16* p = Bt + (size_t)col*ld + k0 + (hi ? 8 : 0);
  h8 lo = *(const h8*)(p);
  h8 hh = *(const h8*)(p + 16);
  v16h b;
#pragma unroll
  for(int i=0;i<8;i++){ b[i]=lo[i]; b[i+8]=hh[i]; }
  return b;
}

DEV v16h load_b_lds(const _Float16* p, int hi){   // p = &row[0], padded row
  const _Float16* q = p + (hi ? 8 : 0);
  h8 lo = *(const h8*)(q);
  h8 hh = *(const h8*)(q + 16);
  v16h b;
#pragma unroll
  for(int i=0;i<8;i++){ b[i]=lo[i]; b[i+8]=hh[i]; }
  return b;
}

DEV v8f wmma16(v16h a, v16h b, v8f c){
  return __builtin_amdgcn_wmma_f32_16x16x32_f16(false, a, false, b, (short)0, c, false, false);
}

DEV size_t bhsd_idx(int row, int col){   // [B*S, H*DK] -> [B,H,S,DK]
  return ((((size_t)(row >> 10))*NH + (col >> 6))*SEQ + (row & (SEQ-1)))*DK + (col & (DK-1));
}
DEV size_t bhds_idx(int row, int col){   // [B*S, H*DK] -> [B,H,DK,S]  (V^T)
  return ((((size_t)(row >> 10))*NH + (col >> 6))*DK + (col & (DK-1)))*SEQ + (row & (SEQ-1));
}

// ============ staged weight GEMM: O = A[4096xK] @ Bt^T (Bt is [N x K]) ============
// block: 256 thr = 8 waves (4M x 2N), block tile 128x128, wave tile 32x64
// B tile (128 cols x 32 k) double-buffered through LDS via async-to-LDS.
// MODE: 0 = f32 out, 1 = f16 relu out, 2 = f16 scatter [B,H,S,DK] (*scale),
//       3 = f16 scatter [B,H,DK,S]
template<int MODE>
__global__ void k_gemm_w(const _Float16* __restrict__ A, const _Float16* __restrict__ Bt,
                         void* __restrict__ Optr, float scale, int N, int K){
  __shared__ _Float16 sB[2][128][40];          // 40-half row pad: conflict-free b128
  const int tid  = threadIdx.x;
  const int lane = tid & 31;
  const int wave = tid >> 5;
  const int hi   = lane >> 4;
  const int lcol = lane & 15;
  const int row0 = blockIdx.y*128 + (wave>>1)*32;
  const int n0b  = blockIdx.x*128;
  const int wn0  = (wave&1)*64;

  v8f acc[2][4];
#pragma unroll
  for(int i=0;i<2;i++)
#pragma unroll
    for(int j=0;j<4;j++) acc[i][j] = (v8f){0.f,0.f,0.f,0.f,0.f,0.f,0.f,0.f};

  // stage k-tile [n0b..n0b+127] x [k0..k0+31] into sB[buf]
  auto stage = [&](int buf, int k0){
#pragma unroll
    for(int c = tid; c < 512; c += 256){
      int row = c >> 2, seg = c & 3;
      async_cp16(Bt + (size_t)(n0b + row)*K + k0 + seg*8, &sB[buf][row][seg*8]);
    }
  };

  stage(0, 0);
  async_wait();
  __syncthreads();
  int cur = 0;

  for(int k0 = 0; k0 < K; k0 += 32){
    if(k0 + 32 < K) stage(cur ^ 1, k0 + 32);

    v16h a0 = load_a_h16(A, K, row0 + lcol,      k0, hi);
    v16h a1 = load_a_h16(A, K, row0 + 16 + lcol, k0, hi);
#pragma unroll
    for(int j=0;j<4;j++){
      v16h b = load_b_lds(&sB[cur][wn0 + j*16 + lcol][0], hi);
      acc[0][j] = wmma16(a0, b, acc[0][j]);
      acc[1][j] = wmma16(a1, b, acc[1][j]);
    }
    async_wait();
    __syncthreads();
    cur ^= 1;
  }

#pragma unroll
  for(int i=0;i<2;i++){
#pragma unroll
    for(int j=0;j<4;j++){
#pragma unroll
      for(int r=0;r<8;r++){
        int m = row0 + i*16 + r + hi*8;
        int n = n0b + wn0 + j*16 + lcol;
        float v = acc[i][j][r];
        if(MODE == 0){
          ((float*)Optr)[(size_t)m*N + n] = v;
        } else if(MODE == 1){
          ((_Float16*)Optr)[(size_t)m*N + n] = (_Float16)fmaxf(v, 0.f);
        } else if(MODE == 2){
          ((_Float16*)Optr)[bhsd_idx(m, n)] = (_Float16)(v*scale);
        } else {
          ((_Float16*)Optr)[bhds_idx(m, n)] = (_Float16)v;
        }
      }
    }
  }
}

// ============ attention kernels (direct loads, 32x32 wave tile) ============
#define GEMM_PROLOG \
  const int lane = threadIdx.x & 31; \
  const int wave = threadIdx.x >> 5; \
  const int hi   = lane >> 4;        \
  const int lcol = lane & 15;        \
  const int row0 = blockIdx.y*128 + (wave>>1)*32; \
  const int col0 = blockIdx.x*64  + (wave&1)*32;  \
  v8f c00 = {0.f,0.f,0.f,0.f,0.f,0.f,0.f,0.f};    \
  v8f c01 = c00, c10 = c00, c11 = c00;

#define MMA4 \
  c00 = wmma16(a0,b0,c00); c01 = wmma16(a0,b1,c01); \
  c10 = wmma16(a1,b0,c10); c11 = wmma16(a1,b1,c11);

// scores[z][q][k] = Q[z] @ K[z]^T  (1/sqrt(dk) folded into Q), pad-mask applied
__global__ void k_scores(const _Float16* __restrict__ qh, const _Float16* __restrict__ kh,
                         const int* __restrict__ tok, float* __restrict__ attnL){
  const int z = blockIdx.z;
  const int b = z >> 3;
  const _Float16* A  = qh + (size_t)z*SEQ*DK;
  const _Float16* Bt = kh + (size_t)z*SEQ*DK;
  GEMM_PROLOG
#pragma unroll
  for(int k0 = 0; k0 < DK; k0 += 32){
    v16h a0 = load_a_h16(A, DK, row0+lcol,    k0, hi);
    v16h a1 = load_a_h16(A, DK, row0+16+lcol, k0, hi);
    v16h b0 = load_b_tr (Bt, DK, k0, col0+lcol,    hi);
    v16h b1 = load_b_tr (Bt, DK, k0, col0+16+lcol, hi);
    MMA4
  }
  float* O = attnL + (size_t)z*SEQ*SEQ;
  const int n0 = col0 + lcol, n1 = n0 + 16;
  const bool p0 = (tok[b*SEQ + n0] == 0);
  const bool p1 = (tok[b*SEQ + n1] == 0);
#pragma unroll
  for(int r=0;r<8;r++){
    int m0 = row0 + r + hi*8, m1 = m0 + 16;
    O[(size_t)m0*SEQ + n0] = p0 ? -1e9f : c00[r];
    O[(size_t)m0*SEQ + n1] = p1 ? -1e9f : c01[r];
    O[(size_t)m1*SEQ + n0] = p0 ? -1e9f : c10[r];
    O[(size_t)m1*SEQ + n1] = p1 ? -1e9f : c11[r];
  }
}

// in-place row softmax over 1024 keys
__global__ void k_softmax(float* __restrict__ attnL){
  __shared__ float red[256];
  const int t = threadIdx.x;
  float* row = attnL + ((size_t)blockIdx.y*SEQ + blockIdx.x)*SEQ;
  float v[4]; float m = -3.4e38f;
#pragma unroll
  for(int i=0;i<4;i++){ v[i] = row[t + i*256]; m = fmaxf(m, v[i]); }
  red[t] = m; __syncthreads();
  for(int s=128;s>0;s>>=1){ if(t<s) red[t] = fmaxf(red[t], red[t+s]); __syncthreads(); }
  m = red[0]; __syncthreads();
  float sum = 0.f;
#pragma unroll
  for(int i=0;i<4;i++){ v[i] = __expf(v[i]-m); sum += v[i]; }
  red[t] = sum; __syncthreads();
  for(int s=128;s>0;s>>=1){ if(t<s) red[t] += red[t+s]; __syncthreads(); }
  float inv = 1.f / red[0];
#pragma unroll
  for(int i=0;i<4;i++) row[t + i*256] = v[i]*inv;
}

// ctx[B,S,H*DK] (f16) = P[z] (f32, SxS) @ V[z]; V stored transposed [B,H,DK,S]
__global__ void k_ctx(const float* __restrict__ attnL, const _Float16* __restrict__ vT,
                      _Float16* __restrict__ ctxh){
  const int z = blockIdx.z;
  const float*    A  = attnL + (size_t)z*SEQ*SEQ;
  const _Float16* Bt = vT    + (size_t)z*DK*SEQ;   // [DK x S] per (b,h)
  GEMM_PROLOG
  for(int k0 = 0; k0 < SEQ; k0 += 32){
    v16h a0 = load_a_f32(A, SEQ, row0+lcol,    k0, hi);
    v16h a1 = load_a_f32(A, SEQ, row0+16+lcol, k0, hi);
    v16h b0 = load_b_tr (Bt, SEQ, k0, col0+lcol,    hi);
    v16h b1 = load_b_tr (Bt, SEQ, k0, col0+16+lcol, hi);
    MMA4
  }
  const int b = z >> 3, h = z & 7;
  const int n0 = col0 + lcol, n1 = n0 + 16;
#pragma unroll
  for(int r=0;r<8;r++){
    int s0 = row0 + r + hi*8;
    size_t base0 = ((size_t)(b*SEQ + s0     ))*DM + h*DK;
    size_t base1 = ((size_t)(b*SEQ + s0 + 16))*DM + h*DK;
    ctxh[base0 + n0] = (_Float16)c00[r];
    ctxh[base0 + n1] = (_Float16)c01[r];
    ctxh[base1 + n0] = (_Float16)c10[r];
    ctxh[base1 + n1] = (_Float16)c11[r];
  }
}

// ---------------- utility kernels ----------------
// transpose-convert: src f32 [K x N] -> dst f16 [N x K], per layer (blockIdx.z)
__global__ void k_cvt_tr(const float* __restrict__ src, _Float16* __restrict__ dst,
                         int K, int N){
  __shared__ _Float16 t[32][33];
  const float* s = src + (size_t)blockIdx.z*K*N;
  _Float16*    d = dst + (size_t)blockIdx.z*N*K;
  int kt = blockIdx.y*32, nt = blockIdx.x*32;
  int tx = threadIdx.x, ty = threadIdx.y;          // 32 x 8
#pragma unroll
  for(int r=0;r<32;r+=8)
    t[ty+r][tx] = (_Float16)s[(size_t)(kt+ty+r)*N + nt+tx];
  __syncthreads();
#pragma unroll
  for(int r=0;r<32;r+=8)
    d[(size_t)(nt+ty+r)*K + kt+tx] = t[tx][ty+r];
}

__global__ void k_copy(const float* __restrict__ s, float* __restrict__ d, int n){
  int i = blockIdx.x*blockDim.x + threadIdx.x;
  if(i < n) d[i] = s[i];
}

// x[b,s,:] = emb[tok] + PE(s,:)
__global__ void k_embed(const int* __restrict__ tok, const float* __restrict__ emb,
                        float* __restrict__ x, _Float16* __restrict__ xh){
  const int row = blockIdx.x;
  const int s   = row & (SEQ-1);
  const int t   = tok[row];
  const float c = -9.210340371976184f / (float)DM;   // -ln(10000)/DM
  for(int d = threadIdx.x; d < DM; d += blockDim.x){
    float freq = __expf((float)(d & ~1) * c);
    float ang  = (float)s * freq;
    float pe   = (d & 1) ? __cosf(ang) : __sinf(ang);
    float v    = emb[(size_t)t*DM + d] + pe;
    x [(size_t)row*DM + d] = v;
    xh[(size_t)row*DM + d] = (_Float16)v;
  }
}

// x = LayerNorm(tmp + x); also writes f16 copy
__global__ void k_ln(const float* __restrict__ tmp, float* __restrict__ x,
                     _Float16* __restrict__ xh){
  __shared__ float r1[256], r2[256];
  const int row = blockIdx.x;
  const int t   = threadIdx.x;
  const float* a = tmp + (size_t)row*DM;
  float* xr = x + (size_t)row*DM;
  float u0 = a[t] + xr[t];
  float u1 = a[t+256] + xr[t+256];
  r1[t] = u0 + u1; r2[t] = u0*u0 + u1*u1; __syncthreads();
  for(int s=128;s>0;s>>=1){ if(t<s){ r1[t]+=r1[t+s]; r2[t]+=r2[t+s]; } __syncthreads(); }
  float mu  = r1[0] * (1.f/DM);
  float var = r2[0] * (1.f/DM) - mu*mu;
  float rs  = rsqrtf(var + 1e-5f);
  float y0 = (u0-mu)*rs, y1 = (u1-mu)*rs;
  xr[t] = y0; xr[t+256] = y1;
  xh[(size_t)row*DM + t]       = (_Float16)y0;
  xh[(size_t)row*DM + t + 256] = (_Float16)y1;
}

// ---------------- host launch ----------------
extern "C" void kernel_launch(void* const* d_in, const int* in_sizes, int n_in,
                              void* d_out, int out_size, void* d_ws, size_t ws_size,
                              hipStream_t stream){
  (void)in_sizes; (void)n_in; (void)out_size; (void)ws_size;
  const int*   tok = (const int*)  d_in[0];
  const float* emb = (const float*)d_in[1];
  const float* Wq  = (const float*)d_in[2];
  const float* Wk  = (const float*)d_in[3];
  const float* Wv  = (const float*)d_in[4];
  const float* Wo  = (const float*)d_in[5];
  const float* W1  = (const float*)d_in[6];
  const float* W2  = (const float*)d_in[7];
  float* out_x    = (float*)d_out;
  float* out_attn = out_x + (size_t)ROWS*DM;

  char* ws = (char*)d_ws; size_t off = 0;
  auto alloc = [&](size_t bytes)->void*{
    void* p = ws + off; off = (off + bytes + 255) & ~(size_t)255; return p; };
  float*    x    = (float*)   alloc((size_t)ROWS*DM*4);
  float*    tmp  = (float*)   alloc((size_t)ROWS*DM*4);
  _Float16* xh   = (_Float16*)alloc((size_t)ROWS*DM*2);
  _Float16* qh   = (_Float16*)alloc((size_t)ROWS*DM*2);
  _Float16* kh   = (_Float16*)alloc((size_t)ROWS*DM*2);
  _Float16* vT   = (_Float16*)alloc((size_t)ROWS*DM*2);
  _Float16* ctxh = (_Float16*)alloc((size_t)ROWS*DM*2);
  _Float16* f1h  = (_Float16*)alloc((size_t)ROWS*FF*2);
  _Float16* Wqt  = (_Float16*)alloc((size_t)NLAY*DM*DM*2);
  _Float16* Wkt  = (_Float16*)alloc((size_t)NLAY*DM*DM*2);
  _Float16* Wvt  = (_Float16*)alloc((size_t)NLAY*DM*DM*2);
  _Float16* Wot  = (_Float16*)alloc((size_t)NLAY*DM*DM*2);
  _Float16* W1t  = (_Float16*)alloc((size_t)NLAY*DM*FF*2);
  _Float16* W2t  = (_Float16*)alloc((size_t)NLAY*FF*DM*2);

  const dim3 blk(256);
  const dim3 tblk(32,8);
  // weight transpose+convert (one-time per launch)
  k_cvt_tr<<<dim3(DM/32, DM/32, NLAY), tblk, 0, stream>>>(Wq, Wqt, DM, DM);
  k_cvt_tr<<<dim3(DM/32, DM/32, NLAY), tblk, 0, stream>>>(Wk, Wkt, DM, DM);
  k_cvt_tr<<<dim3(DM/32, DM/32, NLAY), tblk, 0, stream>>>(Wv, Wvt, DM, DM);
  k_cvt_tr<<<dim3(DM/32, DM/32, NLAY), tblk, 0, stream>>>(Wo, Wot, DM, DM);
  k_cvt_tr<<<dim3(FF/32, DM/32, NLAY), tblk, 0, stream>>>(W1, W1t, DM, FF);
  k_cvt_tr<<<dim3(DM/32, FF/32, NLAY), tblk, 0, stream>>>(W2, W2t, FF, DM);

  k_embed<<<dim3(ROWS), blk, 0, stream>>>(tok, emb, x, xh);

  for(int l = 0; l < NLAY; l++){
    const _Float16* wq = Wqt + (size_t)l*DM*DM;
    const _Float16* wk = Wkt + (size_t)l*DM*DM;
    const _Float16* wv = Wvt + (size_t)l*DM*DM;
    const _Float16* wo = Wot + (size_t)l*DM*DM;
    const _Float16* w1 = W1t + (size_t)l*DM*FF;
    const _Float16* w2 = W2t + (size_t)l*FF*DM;
    float* attnL = out_attn + (size_t)l*BATCH*NH*SEQ*SEQ;

    k_gemm_w<2><<<dim3(DM/128, ROWS/128), blk, 0, stream>>>(xh, wq, qh,  0.125f, DM, DM);
    k_gemm_w<2><<<dim3(DM/128, ROWS/128), blk, 0, stream>>>(xh, wk, kh,  1.0f,   DM, DM);
    k_gemm_w<3><<<dim3(DM/128, ROWS/128), blk, 0, stream>>>(xh, wv, vT,  1.0f,   DM, DM);

    k_scores <<<dim3(SEQ/64, SEQ/128, BATCH*NH), blk, 0, stream>>>(qh, kh, tok, attnL);
    k_softmax<<<dim3(SEQ, BATCH*NH),             blk, 0, stream>>>(attnL);
    k_ctx    <<<dim3(DK/64, SEQ/128, BATCH*NH),  blk, 0, stream>>>(attnL, vT, ctxh);

    k_gemm_w<0><<<dim3(DM/128, ROWS/128), blk, 0, stream>>>(ctxh, wo, tmp, 1.0f, DM, DM);
    k_ln       <<<dim3(ROWS),             blk, 0, stream>>>(tmp, x, xh);
    k_gemm_w<1><<<dim3(FF/128, ROWS/128), blk, 0, stream>>>(xh, w1, f1h, 1.0f, FF, DM);
    k_gemm_w<0><<<dim3(DM/128, ROWS/128), blk, 0, stream>>>(f1h, w2, tmp, 1.0f, DM, FF);
    k_ln       <<<dim3(ROWS),             blk, 0, stream>>>(tmp, x, xh);
  }

  k_copy<<<dim3((ROWS*DM + 255)/256), blk, 0, stream>>>(x, out_x, ROWS*DM);
}
